// AutoregressiveDecoder_61014305407094
// MI455X (gfx1250) — matrix operand order repaired and checked by reading, not compiled, not run on hardware
//
#include <hip/hip_runtime.h>
#include <hip/hip_bf16.h>

// ---------------------------------------------------------------------------
// MADE autoregressive decoder for MI455X (gfx1250, wave32, WMMA).
// Sequential 4095-step scan; per step two GEMMs with batch 2 padded to M=16,
// computed with v_wmma_f32_16x16x32_bf16 over bf16 weights pre-masked and
// pre-swizzled into WMMA B-fragment layout (L2-resident, ~20 MB).
// Persistent kernel, 32 WGs x 256 thr (256 waves), software grid barrier.
// ---------------------------------------------------------------------------

#define Dg   4096
#define Hg   1200
#define HP   1216          // H padded to multiple of 32 (K for phase 2)
#define NT1  75            // 1200/16 column tiles, phase 1
#define KT1  128           // 4096/32 k tiles, phase 1
#define NT2  256           // 4096/16 column tiles, phase 2
#define KT2  38            // 1216/32 k tiles, phase 2
#define NBLK 32            // persistent grid blocks
#define TPB  256

typedef __attribute__((ext_vector_type(16))) __bf16   v16bf;
typedef __attribute__((ext_vector_type(8)))  float    v8f;
typedef __attribute__((ext_vector_type(4)))  unsigned v4u;

union Frag16 { v16bf v; v4u q[2]; };

// ---------------- threefry2x32 (matches JAX key(42)/fold_in/uniform) --------
struct U2 { unsigned x, y; };

__device__ inline U2 threefry2x32(unsigned k0, unsigned k1, unsigned c0, unsigned c1) {
  unsigned ks2 = k0 ^ k1 ^ 0x1BD11BDAu;
  unsigned x0 = c0 + k0, x1 = c1 + k1;
#define TFR(r) { x0 += x1; x1 = (x1 << (r)) | (x1 >> (32 - (r))); x1 ^= x0; }
  TFR(13) TFR(15) TFR(26) TFR(6)
  x0 += k1;  x1 += ks2 + 1u;
  TFR(17) TFR(29) TFR(16) TFR(24)
  x0 += ks2; x1 += k0 + 2u;
  TFR(13) TFR(15) TFR(26) TFR(6)
  x0 += k0;  x1 += k1 + 3u;
  TFR(17) TFR(29) TFR(16) TFR(24)
  x0 += k1;  x1 += ks2 + 4u;
  TFR(13) TFR(15) TFR(26) TFR(6)
  x0 += ks2; x1 += k0 + 5u;
#undef TFR
  U2 r; r.x = x0; r.y = x1; return r;
}

__device__ inline float u01(unsigned b) {
  return __uint_as_float((b >> 9) | 0x3f800000u) - 1.0f;
}

// MADE hidden degree (1-based h index passed as 0-based)
__device__ inline int hidDeg(int h) {
  unsigned a = (unsigned)(h + 1) * 4095u;          // (h+1)*(D-1)
  int v = (int)((a + 1200u) / 1201u);              // ceil(a/(H+1))
  return v < 1 ? 1 : v;
}

// ---------------- encoder stack (tiny, scalar f32) --------------------------
__global__ void k_enc1(const float* __restrict__ z, const float* __restrict__ W,
                       const float* __restrict__ b, float* __restrict__ h1) {
  int t = blockIdx.x * blockDim.x + threadIdx.x;
  if (t >= 2 * Hg) return;
  int bb = t / Hg, hh = t % Hg;
  float s = b[hh];
  for (int l = 0; l < 64; ++l) s += z[bb * 64 + l] * W[l * Hg + hh];
  h1[t] = tanhf(s);
}

__global__ void k_enc2(const float* __restrict__ h1, const float* __restrict__ W,
                       const float* __restrict__ b, float* __restrict__ h2) {
  int t = blockIdx.x * blockDim.x + threadIdx.x;
  if (t >= 2 * Hg) return;
  int bb = t / Hg, hh = t % Hg;
  float s = b[hh];
  for (int l = 0; l < Hg; ++l) s += h1[bb * Hg + l] * W[(size_t)l * Hg + hh];
  h2[t] = tanhf(s);
}

__global__ void k_enc3(const float* __restrict__ h2, const float* __restrict__ W,
                       const float* __restrict__ b, float* __restrict__ pmean) {
  int t = blockIdx.x * blockDim.x + threadIdx.x;   // 8192 threads exactly
  int bb = t / Dg, d = t % Dg;
  float s = b[d];
  for (int l = 0; l < Hg; ++l) s += h2[bb * Hg + l] * W[(size_t)l * Dg + d];
  pmean[t] = s;
}

// ---------------- mask + bf16 + WMMA-B-fragment swizzle ---------------------
// Fragment f covers (n_tile, k_tile); element layout: lane l (0..31), e (0..15)
//   column n = n_tile*16 + (l & 15)
//   row    k = k_tile*32 + ((l >= 16) ? 16 : 0) + e
// stored at Wswz[f*512 + l*16 + e]  -> each lane loads 32 contiguous bytes.
__global__ void k_prep(const float* __restrict__ W_a1, const float* __restrict__ W_a2,
                       __bf16* __restrict__ Wm1, __bf16* __restrict__ Wm2) {
  const size_t N1 = (size_t)NT1 * KT1 * 512;
  const size_t N2 = (size_t)NT2 * KT2 * 512;
  for (size_t idx = blockIdx.x * (size_t)blockDim.x + threadIdx.x;
       idx < N1 + N2; idx += (size_t)gridDim.x * blockDim.x) {
    if (idx < N1) {
      size_t frag = idx >> 9; int r = (int)(idx & 511); int lane = r >> 4; int e = r & 15;
      int nt = (int)(frag / KT1), kt = (int)(frag % KT1);
      int n = nt * 16 + (lane & 15);
      int k = kt * 32 + ((lane >= 16) ? 16 : 0) + e;
      float v = 0.0f;
      if ((k + 1) <= hidDeg(n)) v = W_a1[(size_t)k * Hg + n];  // M1[d=k][h=n]
      Wm1[idx] = (__bf16)v;
    } else {
      size_t j = idx - N1;
      size_t frag = j >> 9; int r = (int)(j & 511); int lane = r >> 4; int e = r & 15;
      int nt = (int)(frag / KT2), kt = (int)(frag % KT2);
      int n = nt * 16 + (lane & 15);
      int k = kt * 32 + ((lane >= 16) ? 16 : 0) + e;
      float v = 0.0f;
      if (k < Hg && hidDeg(k) < (n + 1)) v = W_a2[(size_t)k * Dg + n];  // M2[h=k][d=n]
      Wm2[j] = (__bf16)v;
    }
  }
}

// ---------------- init scratch / barrier / col-0 outputs --------------------
__global__ void k_init(const float* __restrict__ pmean, __bf16* __restrict__ sampleA,
                       __bf16* __restrict__ hbuf, unsigned* __restrict__ bar,
                       float* __restrict__ outp) {
  size_t t = blockIdx.x * (size_t)blockDim.x + threadIdx.x;
  size_t stride = (size_t)gridDim.x * blockDim.x;
  for (size_t i = t; i < (size_t)16 * Dg; i += stride) sampleA[i] = (__bf16)0.0f;
  for (size_t i = t; i < (size_t)16 * HP; i += stride) hbuf[i]   = (__bf16)0.0f;
  if (t == 0) {
    bar[0] = 0u; bar[1] = 0u;
    outp[0]  = pmean[0];        // mean_out[0,0] = param_mean[0,0]
    outp[Dg] = pmean[Dg];       // mean_out[1,0] = param_mean[1,0]
  }
}

// ---------------- device-wide software barrier ------------------------------
__device__ inline void grid_barrier(unsigned* bar, unsigned target) {
  __syncthreads();
  if (threadIdx.x == 0) {
    unsigned old = __hip_atomic_fetch_add(&bar[0], 1u, __ATOMIC_ACQ_REL,
                                          __HIP_MEMORY_SCOPE_AGENT);
    if (old == NBLK - 1) {
      __hip_atomic_store(&bar[0], 0u, __ATOMIC_RELAXED, __HIP_MEMORY_SCOPE_AGENT);
      __hip_atomic_fetch_add(&bar[1], 1u, __ATOMIC_RELEASE, __HIP_MEMORY_SCOPE_AGENT);
    } else {
      while (__hip_atomic_load(&bar[1], __ATOMIC_ACQUIRE,
                               __HIP_MEMORY_SCOPE_AGENT) < target) {
        __builtin_amdgcn_s_sleep(1);
      }
    }
  }
  __syncthreads();
}

// A-fragment loader: 16x32 bf16 A tile from row-major [16][rowStride] source.
// lane l holds row m = l&15; K offsets {16h..16h+15}, h = l>>4.
__device__ inline v16bf load_afrag(const __bf16* __restrict__ src, int rowStride,
                                   int kbase, int lan15, int hv) {
  Frag16 a;
  const __bf16* ar = src + (size_t)lan15 * rowStride + kbase + hv * 16;
  a.q[0] = *(const v4u*)(ar);
  a.q[1] = *(const v4u*)(ar + 8);
  return a.v;
}

// ---------------- persistent autoregressive scan ----------------------------
__global__ void __launch_bounds__(TPB, 1)
k_persist(const float* __restrict__ pmean, const float* __restrict__ b_a1,
          const float* __restrict__ b_a2, const __bf16* __restrict__ Wm1,
          const __bf16* __restrict__ Wm2, __bf16* __restrict__ sampleA,
          __bf16* __restrict__ hbuf, unsigned* __restrict__ bar,
          float* __restrict__ outp) {
  const int gtid  = blockIdx.x * TPB + threadIdx.x;     // 0..8191
  const int wave  = (blockIdx.x << 3) | (threadIdx.x >> 5);  // 0..255
  const int lane  = threadIdx.x & 31;
  const int hv    = lane >> 4;
  const int lan15 = lane & 15;

  // ---- sample0: logits = [param_mean[:, :1], zeros], u = uniform(fold_in(key,0))
  if (gtid < Dg) {
    const int col = gtid;
    U2 fk = threefry2x32(0u, 42u, 0u, 0u);                       // fold_in(key42, 0)
    U2 r  = threefry2x32(fk.x, fk.y, (unsigned)col, (unsigned)(col + Dg));
    float l0 = (col == 0) ? pmean[0]  : 0.0f;
    float l1 = (col == 0) ? pmean[Dg] : 0.0f;
    float s0 = (u01(r.x) < 1.0f / (1.0f + expf(-l0))) ? 1.0f : 0.0f;
    float s1 = (u01(r.y) < 1.0f / (1.0f + expf(-l1))) ? 1.0f : 0.0f;
    sampleA[col]      = (__bf16)s0;
    sampleA[Dg + col] = (__bf16)s1;
  }
  unsigned tgt = 1;
  grid_barrier(bar, tgt++);

  for (int i = 1; i < Dg; ++i) {
    // ---- phase 1: h = tanh(sample @ Wm1 + b_a1), one wave per 16-col tile
    for (int tile = wave; tile < NT1; tile += 256) {
      v8f acc = {};
      const __bf16* wbase = Wm1 + ((size_t)tile * KT1) * 512 + lane * 16;
      for (int kt = 0; kt < KT1; ++kt) {
        v16bf a = load_afrag(sampleA, Dg, kt * 32, lan15, hv);
        v16bf b = *(const v16bf*)(wbase + (size_t)kt * 512);
        acc = __builtin_amdgcn_wmma_f32_16x16x32_bf16(
            false, a, false, b, (short)0, acc, false, false);
      }
      const int col = tile * 16 + lan15;
      const float bias = b_a1[col];
      for (int v = 0; v < 8; ++v) {
        int m = v + hv * 8;
        hbuf[(size_t)m * HP + col] = (__bf16)tanhf(acc[v] + bias);
      }
    }
    grid_barrier(bar, tgt++);

    // ---- phase 2: mean = pmean + h @ Wm2 + b_a2; RNG + sample update
    {
      const int tile = wave;  // exactly 256 tiles, one per wave
      v8f acc = {};
      const __bf16* wbase = Wm2 + ((size_t)tile * KT2) * 512 + lane * 16;
      for (int kt = 0; kt < KT2; ++kt) {
        v16bf a = load_afrag(hbuf, HP, kt * 32, lan15, hv);
        v16bf b = *(const v16bf*)(wbase + (size_t)kt * 512);
        acc = __builtin_amdgcn_wmma_f32_16x16x32_bf16(
            false, a, false, b, (short)0, acc, false, false);
      }
      if (lane < 16) {                    // rows 0,1 live in acc[0], acc[1]
        const int col = tile * 16 + lane;
        float m0 = acc[0] + b_a2[col] + pmean[col];
        float m1 = acc[1] + b_a2[col] + pmean[Dg + col];
        if (col == i) { outp[col] = m0; outp[Dg + col] = m1; }
        U2 fk = threefry2x32(0u, 42u, 0u, (unsigned)i);          // fold_in(key42, i)
        U2 r  = threefry2x32(fk.x, fk.y, (unsigned)col, (unsigned)(col + Dg));
        if (col >= i) {
          float s0 = (u01(r.x) < 1.0f / (1.0f + expf(-m0))) ? 1.0f : 0.0f;
          float s1 = (u01(r.y) < 1.0f / (1.0f + expf(-m1))) ? 1.0f : 0.0f;
          sampleA[col]      = (__bf16)s0;
          sampleA[Dg + col] = (__bf16)s1;
        }
      }
    }
    grid_barrier(bar, tgt++);
  }

  // ---- emit sample_out (f32) after final barrier
  {
    const int b   = gtid >> 12;        // 0..1
    const int col = gtid & (Dg - 1);
    outp[2 * Dg + gtid] = (float)sampleA[(size_t)b * Dg + col];
  }
}

// ---------------------------------------------------------------------------
extern "C" void kernel_launch(void* const* d_in, const int* in_sizes, int n_in,
                              void* d_out, int out_size, void* d_ws, size_t ws_size,
                              hipStream_t stream) {
  const float* z    = (const float*)d_in[0];
  const float* W_d1 = (const float*)d_in[1];
  const float* b_d1 = (const float*)d_in[2];
  const float* W_d2 = (const float*)d_in[3];
  const float* b_d2 = (const float*)d_in[4];
  const float* W_bp = (const float*)d_in[5];
  const float* b_bp = (const float*)d_in[6];
  const float* W_a1 = (const float*)d_in[7];
  const float* b_a1 = (const float*)d_in[8];
  const float* W_a2 = (const float*)d_in[9];
  const float* b_a2 = (const float*)d_in[10];
  float* outp = (float*)d_out;

  char* ws = (char*)d_ws;
  unsigned* bar     = (unsigned*)(ws + 0);           // 2 x u32 (cnt, gen)
  float*    pmean   = (float*)   (ws + 256);         // 2*4096 f32
  float*    h1      = (float*)   (ws + 33024);       // 2*1200 f32
  float*    h2      = (float*)   (ws + 42624);       // 2*1200 f32
  __bf16*   sampleA = (__bf16*)  (ws + 52224);       // 16*4096 bf16 (rows 2..15 zero)
  __bf16*   hbuf    = (__bf16*)  (ws + 183296);      // 16*1216 bf16
  __bf16*   Wm1     = (__bf16*)  (ws + 222208);      // 75*128*512 bf16 swizzled
  __bf16*   Wm2     = (__bf16*)  (ws + 10052608);    // 256*38*512 bf16 swizzled
  // total ws usage ~20.0 MB

  k_enc1<<<(2 * Hg + TPB - 1) / TPB, TPB, 0, stream>>>(z, W_d1, b_d1, h1);
  k_enc2<<<(2 * Hg + TPB - 1) / TPB, TPB, 0, stream>>>(h1, W_d2, b_d2, h2);
  k_enc3<<<(2 * Dg) / TPB, TPB, 0, stream>>>(h2, W_bp, b_bp, pmean);
  k_prep<<<512, TPB, 0, stream>>>(W_a1, W_a2, Wm1, Wm2);
  k_init<<<64, TPB, 0, stream>>>(pmean, sampleA, hbuf, bar, outp);
  k_persist<<<NBLK, TPB, 0, stream>>>(pmean, b_a1, b_a2, Wm1, Wm2,
                                      sampleA, hbuf, bar, outp);
}
